// Attention_66709432042145
// MI455X (gfx1250) — matrix operand order repaired and checked by reading, not compiled, not run on hardware
//
#include <hip/hip_runtime.h>

typedef __bf16 bf16_t;
typedef __attribute__((ext_vector_type(16))) __bf16 v16bf;
typedef __attribute__((ext_vector_type(8)))  __bf16 v8bf;
typedef __attribute__((ext_vector_type(8)))  float  v8f;
typedef unsigned int u32x4 __attribute__((ext_vector_type(4)));
typedef int          i32x8 __attribute__((ext_vector_type(8)));
typedef int          i32x4 __attribute__((ext_vector_type(4)));

#define BATCH   4
#define NHEADS  16
#define DHEAD   64
#define NQTOK   1024
#define NKVTOK  4096
#define INNERD  1024

// ---------------------------------------------------------------------------
// WMMA wrapper: D = A(16x32 bf16) * B(32x16 bf16) + C(16x16 f32)
// ---------------------------------------------------------------------------
__device__ __forceinline__ v8f wmma_bf16(v16bf a, v16bf b, v8f c) {
  return __builtin_amdgcn_wmma_f32_16x16x32_bf16(
      /*neg_a=*/false, a, /*neg_b=*/false, b,
      /*c_mod=*/(short)0, c, /*reuse_a=*/false, /*reuse_b=*/false);
}

// CDNA5 async copy: 16 bytes global -> LDS, tracked by ASYNCcnt (no VGPR data).
__device__ __forceinline__ void async_copy_16B(const bf16_t* gsrc, bf16_t* ldst) {
  unsigned lds_off = (unsigned)(uintptr_t)ldst;   // flat addr low 32 bits == LDS byte offset
  unsigned long long gaddr = (unsigned long long)(uintptr_t)gsrc;
  asm volatile("global_load_async_to_lds_b128 %0, %1, off"
               :: "v"(lds_off), "v"(gaddr) : "memory");
}
__device__ __forceinline__ void wait_async0() {
  asm volatile("s_wait_asynccnt 0" ::: "memory");
}

// ---------------------------------------------------------------------------
// CDNA5 Tensor Data Mover: 2-D tile load global -> LDS (TENSORcnt-tracked).
// D# packing per CDNA5 ISA ch.8: group0 = {flags, lds_addr, global_addr, type=2},
// group1 = {ctrl, tensor dims, tile dims, dim0 stride}; groups 2/3 zero (2-D).
// Pads LDS rows: 32 dwords data (+pad_interval code 4) then 4 dwords pad
// (pad_amount code 3) -> row stride 72 bf16, matching the compute-side layout.
// ---------------------------------------------------------------------------
__device__ __forceinline__ void tdm_load_2d(bf16_t* lds_dst, const bf16_t* gsrc,
                                            unsigned tensor_d0, unsigned tensor_d1,
                                            unsigned tile_d0, unsigned tile_d1,
                                            unsigned stride_d0) {
  unsigned long long ga = (unsigned long long)(uintptr_t)gsrc;
  u32x4 g0;
  g0[0] = 1u;                                        // count=1, user descriptor
  g0[1] = (unsigned)(uintptr_t)lds_dst;              // LDS byte address
  g0[2] = (unsigned)(ga & 0xFFFFFFFFu);              // global_addr[31:0]
  g0[3] = (unsigned)((ga >> 32) & 0x01FFFFFFu) | (2u << 30);  // addr[56:32] | type=2
  i32x8 g1;
  g1[0] = (int)((1u << 16) |                          // data_size = 2 bytes
                (1u << 20) |                          // pad_enable
                (4u << 22) |                          // pad_interval: 32 dwords
                (3u << 25));                          // pad_amount: 4 dwords
  g1[1] = (int)((tensor_d0 & 0xFFFFu) << 16);         // tensor_dim0[15:0]
  g1[2] = (int)((tensor_d0 >> 16) | ((tensor_d1 & 0xFFFFu) << 16));
  g1[3] = (int)((tensor_d1 >> 16) | ((tile_d0 & 0xFFFFu) << 16));
  g1[4] = (int)(tile_d1 & 0xFFFFu);                   // tile_dim1 (tile_dim2 = 0)
  g1[5] = (int)stride_d0;                             // tensor_dim0_stride[31:0]
  g1[6] = 0;
  g1[7] = 0;
  i32x4 gz = {0, 0, 0, 0};
#if __has_include(<hip/amd_detail/amd_gfx1250_TDM.h>)
  i32x8 gz8 = {0, 0, 0, 0, 0, 0, 0, 0};
  __builtin_amdgcn_tensor_load_to_lds(g0, g1, gz, gz, gz8, 0);
#else
  __builtin_amdgcn_tensor_load_to_lds(g0, g1, gz, gz, 0);
#endif
}

// A-fragment: 16x32 bf16 tile from row-major [rows x ld] at (row0, k0).
// Lanes 0-15: M=lane, elems 0..7 -> K=k0..k0+7, elems 8..15 -> K=k0+16..k0+23
// Lanes 16-31: M=lane-16, K offset +8 (per CDNA5 ISA 16-bit A layout).
__device__ __forceinline__ v16bf load_a_frag(const bf16_t* base, int row0, int k0, int ld) {
  int lane = threadIdx.x & 31;
  int hf = lane >> 4;
  int m = lane & 15;
  const bf16_t* p = base + (size_t)(row0 + m) * ld + k0 + hf * 8;
  v8bf lo = *(const v8bf*)p;
  v8bf hi = *(const v8bf*)(p + 16);
  v16bf f;
#pragma unroll
  for (int i = 0; i < 8; ++i) { f[i] = lo[i]; f[i + 8] = hi[i]; }
  return f;
}

// B-fragment: B is 32x16 (KxN); we read from Bt (N x K row-major) at (n0, k0).
// Lanes 0-15: N=lane, elems 0..7 -> K=k0..k0+7, elems 8..15 -> K=k0+16..k0+23; lanes 16-31: N=lane-16, K offset +8.
__device__ __forceinline__ v16bf load_b_frag(const bf16_t* baseT, int n0, int k0, int ld) {
  int lane = threadIdx.x & 31;
  int hf = lane >> 4;
  int n = lane & 15;
  const bf16_t* p = baseT + (size_t)(n0 + n) * ld + k0 + hf * 8;
  v8bf lo = *(const v8bf*)p;
  v8bf hi = *(const v8bf*)(p + 16);
  v16bf f;
#pragma unroll
  for (int i = 0; i < 8; ++i) { f[i] = lo[i]; f[i + 8] = hi[i]; }
  return f;
}

// ---------------------------------------------------------------------------
// Elementwise f32 -> bf16 cast
// ---------------------------------------------------------------------------
__global__ void cast_f32_to_bf16(const float* __restrict__ src,
                                 bf16_t* __restrict__ dst, int n) {
  int i = blockIdx.x * blockDim.x + threadIdx.x;
  int stride = gridDim.x * blockDim.x;
  for (; i < n; i += stride) dst[i] = (bf16_t)src[i];
}

// Transpose + cast: W [K x N] f32 -> Wt [N x K] bf16 (LDS-tiled, coalesced)
__global__ void transpose_cast_bf16(const float* __restrict__ W,
                                    bf16_t* __restrict__ Wt, int K, int N) {
  __shared__ float tile[32][33];
  int n0 = blockIdx.x * 32;
  int k0 = blockIdx.y * 32;
  int tx = threadIdx.x, ty = threadIdx.y; // 32 x 8
#pragma unroll
  for (int i = 0; i < 32; i += 8)
    tile[ty + i][tx] = W[(size_t)(k0 + ty + i) * N + n0 + tx];
  __syncthreads();
#pragma unroll
  for (int i = 0; i < 32; i += 8)
    Wt[(size_t)(n0 + ty + i) * K + k0 + tx] = (bf16_t)tile[tx][ty + i];
}

// ---------------------------------------------------------------------------
// Shared GEMM mainloop: C[128x128] += A[128xK] * Bt[128xK]^T  (bf16, f32 acc)
// Block = 256 threads = 8 waves (4x2 wave grid, 32x64 per wave).
// Double-buffered LDS tiles (128x32, padded ld=40), filled by async-to-LDS.
// ---------------------------------------------------------------------------
__device__ __forceinline__ void gemm_issue_tile(const bf16_t* __restrict__ G,
                                                int ldg, int row0, int k0,
                                                bf16_t* __restrict__ lds) {
  int tid = threadIdx.x;
  int row = tid >> 1;             // 0..127
  int col = (tid & 1) << 4;       // 0 or 16
  const bf16_t* src = G + (size_t)(row0 + row) * ldg + k0 + col;
  bf16_t* dst = lds + row * 40 + col;
  async_copy_16B(src, dst);
  async_copy_16B(src + 8, dst + 8);
}

__device__ __forceinline__ void gemm_mainloop(const bf16_t* __restrict__ A,
                                              const bf16_t* __restrict__ Bt,
                                              int K,
                                              bf16_t (*ldsA)[128 * 40],
                                              bf16_t (*ldsB)[128 * 40],
                                              v8f acc[2][4]) {
  int m_block = blockIdx.y * 128;
  int n_block = blockIdx.x * 128;
  int w = threadIdx.x >> 5;
  int wr = w >> 1;   // 0..3
  int wc = w & 1;    // 0..1

  v8f z = {};
#pragma unroll
  for (int i = 0; i < 2; ++i)
#pragma unroll
    for (int j = 0; j < 4; ++j) acc[i][j] = z;

  gemm_issue_tile(A, K, m_block, 0, ldsA[0]);
  gemm_issue_tile(Bt, K, n_block, 0, ldsB[0]);

  int nchunks = K >> 5;
  for (int kc = 0; kc < nchunks; ++kc) {
    wait_async0();        // this wave's async tile(s) have landed in LDS
    __syncthreads();      // everyone's have
    if (kc + 1 < nchunks) {
      gemm_issue_tile(A, K, m_block, (kc + 1) << 5, ldsA[(kc + 1) & 1]);
      gemm_issue_tile(Bt, K, n_block, (kc + 1) << 5, ldsB[(kc + 1) & 1]);
    }
    const bf16_t* cA = ldsA[kc & 1];
    const bf16_t* cB = ldsB[kc & 1];
    v16bf a0 = load_a_frag(cA, wr * 32, 0, 40);
    v16bf a1 = load_a_frag(cA, wr * 32 + 16, 0, 40);
#pragma unroll
    for (int j = 0; j < 4; ++j) {
      v16bf bj = load_b_frag(cB, wc * 64 + j * 16, 0, 40);
      acc[0][j] = wmma_bf16(a0, bj, acc[0][j]);
      acc[1][j] = wmma_bf16(a1, bj, acc[1][j]);
    }
  }
}

// Q projection: q = x @ Wq  -> scatter to [B][H][NQ][64] bf16
__global__ __launch_bounds__(256) void gemm_qproj(const bf16_t* __restrict__ A,
                                                  const bf16_t* __restrict__ Bt,
                                                  bf16_t* __restrict__ qbuf) {
  __shared__ bf16_t ldsA[2][128 * 40];
  __shared__ bf16_t ldsB[2][128 * 40];
  v8f acc[2][4];
  gemm_mainloop(A, Bt, INNERD, ldsA, ldsB, acc);

  int w = threadIdx.x >> 5, wr = w >> 1, wc = w & 1;
  int lane = threadIdx.x & 31, hf = lane >> 4, ln = lane & 15;
  int gm_base = blockIdx.y * 128 + wr * 32;
  int gn_base = blockIdx.x * 128 + wc * 64;
#pragma unroll
  for (int i = 0; i < 2; ++i)
#pragma unroll
    for (int j = 0; j < 4; ++j)
#pragma unroll
      for (int r = 0; r < 8; ++r) {
        int gm = gm_base + i * 16 + hf * 8 + r;
        int gn = gn_base + j * 16 + ln;
        int b = gm >> 10, qi = gm & 1023;
        int h = gn >> 6, d = gn & 63;
        qbuf[(((size_t)(b * NHEADS + h)) * NQTOK + qi) * DHEAD + d] =
            (bf16_t)acc[i][j][r];
      }
}

// KV projection: kv = context @ Wkv -> K to [B][H][NKV][64], V to Vt [B][H][64][NKV]
__global__ __launch_bounds__(256) void gemm_kvproj(const bf16_t* __restrict__ A,
                                                   const bf16_t* __restrict__ Bt,
                                                   bf16_t* __restrict__ kbuf,
                                                   bf16_t* __restrict__ vtbuf) {
  __shared__ bf16_t ldsA[2][128 * 40];
  __shared__ bf16_t ldsB[2][128 * 40];
  v8f acc[2][4];
  gemm_mainloop(A, Bt, INNERD, ldsA, ldsB, acc);

  int w = threadIdx.x >> 5, wr = w >> 1, wc = w & 1;
  int lane = threadIdx.x & 31, hf = lane >> 4, ln = lane & 15;
  int gm_base = blockIdx.y * 128 + wr * 32;
  int gn_base = blockIdx.x * 128 + wc * 64;
#pragma unroll
  for (int i = 0; i < 2; ++i)
#pragma unroll
    for (int j = 0; j < 4; ++j)
#pragma unroll
      for (int r = 0; r < 8; ++r) {
        int gm = gm_base + i * 16 + hf * 8 + r;
        int gn = gn_base + j * 16 + ln;
        int b = gm >> 12, kvi = gm & 4095;
        bf16_t v = (bf16_t)acc[i][j][r];
        if (gn < INNERD) {
          int h = gn >> 6, d = gn & 63;
          kbuf[(((size_t)(b * NHEADS + h)) * NKVTOK + kvi) * DHEAD + d] = v;
        } else {
          int c = gn - INNERD;
          int h = c >> 6, d = c & 63;
          vtbuf[(((size_t)(b * NHEADS + h)) * DHEAD + d) * NKVTOK + kvi] = v;
        }
      }
}

// O projection: out = attn_out @ Wo + bo  (f32 out, row-major [B*NQ, 1024])
__global__ __launch_bounds__(256) void gemm_oproj(const bf16_t* __restrict__ A,
                                                  const bf16_t* __restrict__ Bt,
                                                  const float* __restrict__ bo,
                                                  float* __restrict__ out) {
  __shared__ bf16_t ldsA[2][128 * 40];
  __shared__ bf16_t ldsB[2][128 * 40];
  v8f acc[2][4];
  gemm_mainloop(A, Bt, INNERD, ldsA, ldsB, acc);

  int w = threadIdx.x >> 5, wr = w >> 1, wc = w & 1;
  int lane = threadIdx.x & 31, hf = lane >> 4, ln = lane & 15;
  int gm_base = blockIdx.y * 128 + wr * 32;
  int gn_base = blockIdx.x * 128 + wc * 64;
#pragma unroll
  for (int i = 0; i < 2; ++i)
#pragma unroll
    for (int j = 0; j < 4; ++j) {
      int gn = gn_base + j * 16 + ln;
      float bv = bo[gn];
#pragma unroll
      for (int r = 0; r < 8; ++r) {
        int gm = gm_base + i * 16 + hf * 8 + r;
        out[(size_t)gm * INNERD + gn] = acc[i][j][r] + bv;
      }
    }
}

// ---------------------------------------------------------------------------
// Flash attention: block = (128 q-rows, one (b,h)); 8 waves x 16 rows each.
// K/Vt tiles (64 kv cols) staged by the Tensor Data Mover (double buffered);
// online softmax; P@V through per-wave LDS staging.
// ---------------------------------------------------------------------------
__device__ __forceinline__ void attn_issue_tiles(int w,
                                                 const bf16_t* __restrict__ Kp,
                                                 const bf16_t* __restrict__ Vp,
                                                 int kv0,
                                                 bf16_t* __restrict__ bufK,
                                                 bf16_t* __restrict__ bufV) {
  if (w == 0) {
    // K tile: tensor [NKV rows x 64 cols], tile 64x64 starting at row kv0
    tdm_load_2d(bufK, Kp + (size_t)kv0 * DHEAD,
                /*tensor_d0=*/DHEAD, /*tensor_d1=*/NKVTOK,
                /*tile_d0=*/64, /*tile_d1=*/64, /*stride_d0=*/DHEAD);
  } else if (w == 1) {
    // Vt tile: tensor [64 rows x NKV cols], tile 64x64 starting at col kv0
    tdm_load_2d(bufV, Vp + kv0,
                /*tensor_d0=*/NKVTOK, /*tensor_d1=*/DHEAD,
                /*tile_d0=*/64, /*tile_d1=*/64, /*stride_d0=*/NKVTOK);
  }
}

__global__ __launch_bounds__(256) void attn_flash(const bf16_t* __restrict__ qbuf,
                                                  const bf16_t* __restrict__ kbuf,
                                                  const bf16_t* __restrict__ vtbuf,
                                                  const float* __restrict__ bias,
                                                  const unsigned char* __restrict__ mask,
                                                  bf16_t* __restrict__ attb) {
  __shared__ bf16_t ldsK[2][64 * 72];   // K tile: [kv row][d], TDM-padded ld=72
  __shared__ bf16_t ldsV[2][64 * 72];   // Vt tile: [d row][kv], TDM-padded ld=72
  __shared__ bf16_t ldsP[8][16 * 72];   // per-wave P staging (16 x 64)

  int bh = blockIdx.y;
  int b = bh >> 4;
  int h = bh & 15;
  int w = threadIdx.x >> 5;
  int lane = threadIdx.x & 31;
  int hf = lane >> 4;
  int ln = lane & 15;
  int q0 = blockIdx.x * 128 + w * 16;

  const bf16_t* Q  = qbuf  + (size_t)bh * NQTOK * DHEAD;
  const bf16_t* Kp = kbuf  + (size_t)bh * NKVTOK * DHEAD;
  const bf16_t* Vp = vtbuf + (size_t)bh * DHEAD * NKVTOK;
  const float*  Bp = bias  + (size_t)b * NQTOK * NKVTOK;
  const unsigned char* Mp = mask + (size_t)b * NKVTOK;

  // Q fragments for this wave's 16 rows (K = 64 -> 2 frags of 32)
  v16bf qa0 = load_a_frag(Q, q0, 0, DHEAD);
  v16bf qa1 = load_a_frag(Q, q0, 32, DHEAD);

  v8f o[4];
  v8f z = {};
#pragma unroll
  for (int j = 0; j < 4; ++j) o[j] = z;
  float mrow[8], lrow[8];
#pragma unroll
  for (int r = 0; r < 8; ++r) { mrow[r] = -3.0e38f; lrow[r] = 0.0f; }

  const float NEGMAX = -3.402823466e38f;

  attn_issue_tiles(w, Kp, Vp, 0, ldsK[0], ldsV[0]);
  const int NT = NKVTOK / 64;
  for (int t = 0; t < NT; ++t) {
    if (w < 2) __builtin_amdgcn_s_wait_tensorcnt(0);  // issuing waves drain TDM
    __syncthreads();                                  // tile published to block
    if (t + 1 < NT)
      attn_issue_tiles(w, Kp, Vp, (t + 1) * 64, ldsK[(t + 1) & 1], ldsV[(t + 1) & 1]);
    const bf16_t* cK = ldsK[t & 1];
    const bf16_t* cV = ldsV[t & 1];
    int kv0 = t * 64;

    // sim tile: 16 x 64 (4 accumulators), QK^T with K as Bt
    v8f s[4];
#pragma unroll
    for (int j = 0; j < 4; ++j) {
      s[j] = z;
      v16bf b0 = load_b_frag(cK, j * 16, 0, 72);
      s[j] = wmma_bf16(qa0, b0, s[j]);
      v16bf b1 = load_b_frag(cK, j * 16, 32, 72);
      s[j] = wmma_bf16(qa1, b1, s[j]);
    }

    // scale + bias + mask (+ prefetch next tile's bias rows)
#pragma unroll
    for (int j = 0; j < 4; ++j) {
      int col = kv0 + j * 16 + ln;
      bool mv = Mp[col] != 0;
#pragma unroll
      for (int r = 0; r < 8; ++r) {
        int qrow = q0 + hf * 8 + r;
        const float* bp = &Bp[(size_t)qrow * NKVTOK + col];
        float v = s[j][r] * 0.125f + *bp;
        if (j == 0 && kv0 + 64 < NKVTOK) __builtin_prefetch(bp + 64, 0, 1);
        s[j][r] = mv ? v : NEGMAX;
      }
    }

    // online softmax: row-max / correction / exp / row-sum
    float corr[8];
#pragma unroll
    for (int r = 0; r < 8; ++r) {
      float tmax = fmaxf(fmaxf(s[0][r], s[1][r]), fmaxf(s[2][r], s[3][r]));
#pragma unroll
      for (int xm = 1; xm < 16; xm <<= 1) tmax = fmaxf(tmax, __shfl_xor(tmax, xm));
      float mn = fmaxf(mrow[r], tmax);
      corr[r] = __expf(mrow[r] - mn);
      mrow[r] = mn;
    }
#pragma unroll
    for (int r = 0; r < 8; ++r) {
      float rs = 0.0f;
#pragma unroll
      for (int j = 0; j < 4; ++j) {
        float p = __expf(s[j][r] - mrow[r]);
        s[j][r] = p;
        rs += p;
      }
#pragma unroll
      for (int xm = 1; xm < 16; xm <<= 1) rs += __shfl_xor(rs, xm);
      lrow[r] = lrow[r] * corr[r] + rs;
    }
#pragma unroll
    for (int j = 0; j < 4; ++j)
#pragma unroll
      for (int r = 0; r < 8; ++r) o[j][r] *= corr[r];

    // stage P (D-layout -> row-major bf16) through per-wave LDS
    bf16_t* Pw = &ldsP[w][0];
#pragma unroll
    for (int j = 0; j < 4; ++j)
#pragma unroll
      for (int r = 0; r < 8; ++r)
        Pw[(hf * 8 + r) * 72 + j * 16 + ln] = (bf16_t)s[j][r];
    asm volatile("s_wait_dscnt 0" ::: "memory");

    // O += P @ V   (Vt tile is Bt layout directly)
    v16bf pa0 = load_a_frag(Pw, 0, 0, 72);
    v16bf pa1 = load_a_frag(Pw, 0, 32, 72);
#pragma unroll
    for (int j = 0; j < 4; ++j) {
      v16bf vb0 = load_b_frag(cV, j * 16, 0, 72);
      o[j] = wmma_bf16(pa0, vb0, o[j]);
      v16bf vb1 = load_b_frag(cV, j * 16, 32, 72);
      o[j] = wmma_bf16(pa1, vb1, o[j]);
    }
  }

  // normalize and write attn_out in [B*NQ, INNER] row-major bf16
#pragma unroll
  for (int r = 0; r < 8; ++r) {
    float inv = 1.0f / lrow[r];
    int qrow = q0 + hf * 8 + r;
    size_t rowb = (size_t)(b * NQTOK + qrow) * INNERD;
#pragma unroll
    for (int j = 0; j < 4; ++j)
      attb[rowb + h * 64 + j * 16 + ln] = (bf16_t)(o[j][r] * inv);
  }
}

// ---------------------------------------------------------------------------
// Launch
// ---------------------------------------------------------------------------
extern "C" void kernel_launch(void* const* d_in, const int* in_sizes, int n_in,
                              void* d_out, int out_size, void* d_ws, size_t ws_size,
                              hipStream_t stream) {
  const float* x    = (const float*)d_in[0];
  const float* ctx  = (const float*)d_in[1];
  const unsigned char* mask = (const unsigned char*)d_in[2];
  const float* sbias = (const float*)d_in[3];
  const float* Wq   = (const float*)d_in[4];
  const float* Wkv  = (const float*)d_in[5];
  const float* Wo   = (const float*)d_in[6];
  const float* bo   = (const float*)d_in[7];
  float* out = (float*)d_out;

  char* ws = (char*)d_ws;
  const size_t MB = 1024 * 1024;
  bf16_t* xb   = (bf16_t*)(ws + 0);        //  8 MB: x bf16 [4096,1024]
  bf16_t* cb   = (bf16_t*)(ws + 8 * MB);   // 32 MB: context bf16 [16384,1024]
  bf16_t* Wqt  = (bf16_t*)(ws + 40 * MB);  //  2 MB: Wq^T  [1024,1024]
  bf16_t* Wkvt = (bf16_t*)(ws + 42 * MB);  //  4 MB: Wkv^T [2048,1024]
  bf16_t* Wot  = (bf16_t*)(ws + 46 * MB);  //  2 MB: Wo^T  [1024,1024]
  bf16_t* qbuf = (bf16_t*)(ws + 48 * MB);  //  8 MB: Q [B,H,NQ,64]
  bf16_t* kbuf = (bf16_t*)(ws + 56 * MB);  // 32 MB: K [B,H,NKV,64]
  bf16_t* vtb  = (bf16_t*)(ws + 88 * MB);  // 32 MB: V^T [B,H,64,NKV]
  bf16_t* attb = (bf16_t*)(ws + 120 * MB); //  8 MB: attn out [B*NQ,1024]

  // 1) casts
  {
    int n1 = BATCH * NQTOK * INNERD;
    cast_f32_to_bf16<<<dim3((n1 + 2047) / 2048), dim3(256), 0, stream>>>(x, xb, n1);
    int n2 = BATCH * NKVTOK * INNERD;
    cast_f32_to_bf16<<<dim3((n2 + 2047) / 2048), dim3(256), 0, stream>>>(ctx, cb, n2);
  }
  // 2) weight transposes (to B^T layout for contiguous B-fragments)
  transpose_cast_bf16<<<dim3(1024 / 32, 1024 / 32), dim3(32, 8), 0, stream>>>(Wq, Wqt, 1024, 1024);
  transpose_cast_bf16<<<dim3(2048 / 32, 1024 / 32), dim3(32, 8), 0, stream>>>(Wkv, Wkvt, 1024, 2048);
  transpose_cast_bf16<<<dim3(1024 / 32, 1024 / 32), dim3(32, 8), 0, stream>>>(Wo, Wot, 1024, 1024);

  // 3) Q projection: [4096,1024] x [1024,1024]
  gemm_qproj<<<dim3(1024 / 128, 4096 / 128), dim3(256), 0, stream>>>(xb, Wqt, qbuf);
  // 4) KV projection: [16384,1024] x [1024,2048]
  gemm_kvproj<<<dim3(2048 / 128, 16384 / 128), dim3(256), 0, stream>>>(cb, Wkvt, kbuf, vtb);
  // 5) flash attention: grid (NQ/128, B*H)
  attn_flash<<<dim3(NQTOK / 128, BATCH * NHEADS), dim3(256), 0, stream>>>(
      qbuf, kbuf, vtb, sbias, mask, attb);
  // 6) output projection + bias: [4096,1024] x [1024,1024] -> f32 out
  gemm_oproj<<<dim3(1024 / 128, 4096 / 128), dim3(256), 0, stream>>>(attb, Wot, bo, out);
}
